// NLGS_21122649162349
// MI455X (gfx1250) — compile-verified, matrix operand order checked by reading, not run on hardware
//
#include <hip/hip_runtime.h>
#include <cmath>

typedef _Float16 half_t;
typedef __attribute__((ext_vector_type(2)))  _Float16 h2;
typedef __attribute__((ext_vector_type(16))) _Float16 v16h;
typedef __attribute__((ext_vector_type(8)))  float    v8f;

#define NPTS   262144
#define LOG2T  19
#define TSIZE  (1u << LOG2T)

// packed-weight offsets (in halfs) inside d_ws
#define OFF_V0 0        // Wv0: 16 kt x 4 nt -> 32768 halfs
#define OFF_V1 32768    // Wv1:  2 kt x 4 nt ->  4096 halfs
#define OFF_V2 36864    // Wv2:  2 kt x 1 nt ->  1024 halfs
#define OFF_S0 37888    // Ws0:  2 kt x 4 nt ->  4096 halfs
#define OFF_S1 41984    // Ws1:  2 kt x 1 nt ->  1024 halfs
// total 43008 halfs = 86016 bytes of workspace

struct ResCfg { int r[16]; };

// ---------------------------------------------------------------------------
// Pre-swizzle f32 weights into per-lane WMMA B-fragment order (f16).
// Fragment layout (16-bit B, 32x16, wave32): lane -> N = nt*16 + (lane&15),
// K-half = lane>>4; VGPR v holds K = kt*32 + (v<4?0:16) + 8*(lane>>4) + 2*(v&3) {+0,+1}.
// Rows k < hklim are scaled by 1/256 (h features are stored as 256*h in f16).
// ---------------------------------------------------------------------------
__global__ void ngp_pack_w(const float* __restrict__ W, half_t* __restrict__ out,
                           int Kreal, int Nreal, int KT, int NT, int hklim)
{
    int idx = blockIdx.x * 256 + threadIdx.x;
    int total = KT * NT * 512;
    if (idx >= total) return;
    int e    = idx & 15;
    int lane = (idx >> 4) & 31;
    int rest = idx >> 9;
    int nt = rest % NT;
    int kt = rest / NT;
    int v  = e >> 1, hh = e & 1;
    int lh = lane >> 4;
    int n  = nt * 16 + (lane & 15);
    int k  = kt * 32 + ((v < 4) ? 0 : 16) + 8 * lh + 2 * (v & 3) + hh;
    float val = 0.f;
    if (k < Kreal && n < Nreal) {
        val = W[k * Nreal + n];
        if (k < hklim) val *= (1.0f / 256.0f);
    }
    out[idx] = (half_t)val;
}

// ---------------------------------------------------------------------------
// Fused NGP forward: hash-grid + SH encode -> fused-feature color MLP (WMMA)
// -> sigma MLP (WMMA). 8 waves/block, one 16-point M-tile per wave.
// ---------------------------------------------------------------------------
__global__ __launch_bounds__(256) void ngp_fused_kernel(
    const float* __restrict__ xin, const float* __restrict__ din,
    const float* __restrict__ table, const int* __restrict__ degp,
    const half_t* __restrict__ wsh, float* __restrict__ out, ResCfg rc)
{
    __shared__ half_t hsh[8][16][64];   // per wave: [point][ 32 h*256 | 16 sh | 16 zero ]
    __shared__ half_t hid[8][16][64];   // per wave: hidden activations (f16)

    const int tid  = threadIdx.x;
    const int wave = tid >> 5;
    const int lane = tid & 31;
    const int p    = lane & 15;
    const int hlf  = lane >> 4;
    const int gbase = blockIdx.x * 128 + wave * 16;
    const int gpt   = gbase + p;

    __builtin_prefetch(wsh + (tid << 6), 0, 0);   // warm packed weights (L2)

    // ---------------- Phase 1: encodings -> LDS ----------------
    {
        const float px = xin[gpt*3+0], py = xin[gpt*3+1], pz = xin[gpt*3+2];
        #pragma unroll
        for (int li = 0; li < 8; ++li) {
            const int l   = hlf * 8 + li;     // lanes 0-15: levels 0-7; 16-31: levels 8-15
            const int res = rc.r[l];
            const float fr = (float)res;
            float posx = (px + 1.f) * 0.5f * fr;
            float posy = (py + 1.f) * 0.5f * fr;
            float posz = (pz + 1.f) * 0.5f * fr;
            float fx = floorf(posx), fy = floorf(posy), fz = floorf(posz);
            float wx = posx - fx, wy = posy - fy, wz = posz - fz;
            unsigned cx0 = (unsigned)(int)fx, cy0 = (unsigned)(int)fy, cz0 = (unsigned)(int)fz;
            const long long rp1 = (long long)(res + 1);
            const bool hashed = (rp1 * rp1 * rp1 > (long long)TSIZE);
            const unsigned s = (unsigned)(res + 1);
            float f0 = 0.f, f1 = 0.f;
            #pragma unroll
            for (int c = 0; c < 8; ++c) {
                unsigned bx = (c >> 2) & 1, by = (c >> 1) & 1, bz = c & 1;
                unsigned cx = cx0 + bx, cy = cy0 + by, cz = cz0 + bz;
                unsigned idx;
                if (hashed) {
                    idx = (cx * 1u) ^ (cy * 2654435761u) ^ (cz * 805459861u);
                    idx &= (TSIZE - 1u);
                } else {
                    unsigned ur = (unsigned)res;
                    cx = cx > ur ? ur : cx; cy = cy > ur ? ur : cy; cz = cz > ur ? ur : cz;
                    idx = cx + cy * s + cz * s * s;
                }
                const float2 tv = *(const float2*)(table + (((size_t)l << LOG2T) + idx) * 2);
                float wt = (bx ? wx : 1.f - wx) * (by ? wy : 1.f - wy) * (bz ? wz : 1.f - wz);
                f0 += wt * tv.x; f1 += wt * tv.y;
            }
            hsh[wave][p][2*l+0] = (half_t)(f0 * 256.f);   // scaled; weights carry 1/256
            hsh[wave][p][2*l+1] = (half_t)(f1 * 256.f);
        }
        // Spherical harmonics (degree 4)
        const float dx = din[gpt*3+0], dy = din[gpt*3+1], dz = din[gpt*3+2];
        const int deg = *degp;
        const float xx = dx*dx, yy = dy*dy, zz = dz*dz;
        const float xy = dx*dy, yz = dy*dz, xz = dx*dz;
        auto put = [&](int q, float v) {
            if (deg < 4 && q == deg * deg) v = 0.f;
            hsh[wave][p][32 + q] = (half_t)v;
        };
        if (hlf == 0) {
            put(0,  0.28209479177387814f);
            put(1, -0.48860251190291987f * dy);
            put(2,  0.48860251190291987f * dz);
            put(3, -0.48860251190291987f * dx);
            put(4,  1.0925484305920792f  * xy);
            put(5, -1.0925484305920792f  * yz);
            put(6,  0.94617469575755997f * zz - 0.31539156525251999f);
            put(7, -1.0925484305920792f  * xz);
        } else {
            put(8,  0.54627421529603959f * (xx - yy));
            put(9,  0.59004358992664352f * dy * (3.f*xx - yy));
            put(10, 2.8906114426405538f  * xy * dz);
            put(11, 0.45704579946446572f * dy * (1.f - 5.f*zz));
            put(12, 0.3731763325901154f  * dz * (5.f*zz - 3.f));
            put(13, 0.45704579946446572f * dx * (1.f - 5.f*zz));
            put(14, 1.4453057213202769f  * dz * (xx - yy));
            put(15, 0.59004358992664352f * dx * (xx - 3.f*yy));
        }
        #pragma unroll
        for (int i = 0; i < 8; ++i) hsh[wave][p][48 + 8*hlf + i] = (half_t)0.f; // K pad
    }
    __syncthreads();

    // ---------------- Phase 2: per-lane register copies for feat build ----------------
    const int m = lane & 15;               // A-fragment row this lane owns
    h2 hp[16]; h2 sp4[4];
    {
        const h2* rowp = (const h2*)&hsh[wave][m][0];
        #pragma unroll
        for (int i = 0; i < 16; ++i) hp[i] = rowp[i];            // h*256 (32 halfs)
        #pragma unroll
        for (int i = 0; i < 4; ++i)  sp4[i] = rowp[16 + 4*hlf + i]; // the 4 sh pairs this half uses
    }

    // A-fragment builder from an LDS f16 row (16-bit A 16x32 layout)
    auto ldsA = [&](const half_t* rowbase, int kt) -> v16h {
        v16h a;
        #pragma unroll
        for (int v = 0; v < 8; ++v) {
            const int k = kt*32 + ((v < 4) ? 0 : 16) + 8*hlf + 2*(v & 3);
            h2 pr = *(const h2*)(rowbase + k);
            a[2*v] = pr[0]; a[2*v+1] = pr[1];
        }
        return a;
    };

    // ---------------- GEMM1: feat[16,512] @ Wv0 -> h1[16,64] ----------------
    // feat[m, i*16+j] = h[m,i]*sh[m,j]; built per fragment with one pk_mul per VGPR.
    const v16h* pv0 = (const v16h*)(wsh + OFF_V0);
    v8f acc[4] = {};
    #pragma unroll
    for (int kt = 0; kt < 16; ++kt) {
        v16h a;
        #pragma unroll
        for (int v = 0; v < 8; ++v) {
            const int i = kt*2 + ((v < 4) ? 0 : 1);   // h index (hlf-independent)
            half_t hv = hp[i >> 1][i & 1];
            h2 pr; pr[0] = hv; pr[1] = hv;
            pr = pr * sp4[v & 3];                     // sh pair index = 4*hlf + (v&3)
            a[2*v] = pr[0]; a[2*v+1] = pr[1];
        }
        #pragma unroll
        for (int nt = 0; nt < 4; ++nt) {
            v16h b = pv0[(kt*4 + nt)*32 + lane];
            acc[nt] = __builtin_amdgcn_wmma_f32_16x16x32_f16(false, a, false, b,
                                                             (short)0, acc[nt], false, false);
        }
    }
    #pragma unroll
    for (int nt = 0; nt < 4; ++nt)
        #pragma unroll
        for (int r = 0; r < 8; ++r) {
            float vv = acc[nt][r]; vv = vv > 0.f ? vv : 0.f;          // relu
            hid[wave][r + 8*hlf][nt*16 + m] = (half_t)vv;             // D: lane=N, vgpr=M
        }
    __syncthreads();

    // ---------------- GEMM2: h1 @ Wv1 -> h2[16,64] ----------------
    const v16h* pv1 = (const v16h*)(wsh + OFF_V1);
    v8f acc2[4] = {};
    #pragma unroll
    for (int kt = 0; kt < 2; ++kt) {
        v16h a = ldsA(&hid[wave][m][0], kt);
        #pragma unroll
        for (int nt = 0; nt < 4; ++nt) {
            v16h b = pv1[(kt*4 + nt)*32 + lane];
            acc2[nt] = __builtin_amdgcn_wmma_f32_16x16x32_f16(false, a, false, b,
                                                              (short)0, acc2[nt], false, false);
        }
    }
    __syncthreads();
    #pragma unroll
    for (int nt = 0; nt < 4; ++nt)
        #pragma unroll
        for (int r = 0; r < 8; ++r) {
            float vv = acc2[nt][r]; vv = vv > 0.f ? vv : 0.f;
            hid[wave][r + 8*hlf][nt*16 + m] = (half_t)vv;
        }
    __syncthreads();

    // ---------------- GEMM3: h2 @ Wv2(pad 3->16) -> color ----------------
    const v16h* pv2 = (const v16h*)(wsh + OFF_V2);
    v8f acc3 = {};
    #pragma unroll
    for (int kt = 0; kt < 2; ++kt) {
        v16h a = ldsA(&hid[wave][m][0], kt);
        v16h b = pv2[kt*32 + lane];
        acc3 = __builtin_amdgcn_wmma_f32_16x16x32_f16(false, a, false, b,
                                                      (short)0, acc3, false, false);
    }

    // ---------------- Sigma GEMM1: [h*256 | sh0..7 | 0pad] @ Ws0 -> s1[16,64] ----------------
    const v16h* ps0 = (const v16h*)(wsh + OFF_S0);
    v8f accs[4] = {};
    #pragma unroll
    for (int kt = 0; kt < 2; ++kt) {
        v16h a = ldsA(&hsh[wave][m][0], kt);
        #pragma unroll
        for (int nt = 0; nt < 4; ++nt) {
            v16h b = ps0[(kt*4 + nt)*32 + lane];
            accs[nt] = __builtin_amdgcn_wmma_f32_16x16x32_f16(false, a, false, b,
                                                              (short)0, accs[nt], false, false);
        }
    }
    __syncthreads();
    #pragma unroll
    for (int nt = 0; nt < 4; ++nt)
        #pragma unroll
        for (int r = 0; r < 8; ++r) {
            float vv = accs[nt][r]; vv = vv > 0.f ? vv : 0.f;
            hid[wave][r + 8*hlf][nt*16 + m] = (half_t)vv;
        }
    __syncthreads();

    // ---------------- Sigma GEMM2: s1 @ Ws1(pad 1->16) -> opticity ----------------
    const v16h* ps1 = (const v16h*)(wsh + OFF_S1);
    v8f accw = {};
    #pragma unroll
    for (int kt = 0; kt < 2; ++kt) {
        v16h a = ldsA(&hid[wave][m][0], kt);
        v16h b = ps1[kt*32 + lane];
        accw = __builtin_amdgcn_wmma_f32_16x16x32_f16(false, a, false, b,
                                                      (short)0, accw, false, false);
    }

    // ---------------- outputs: sigmoid + store ----------------
    if (m < 3) {                 // color channel m held by lanes {m, 16+m}
        #pragma unroll
        for (int r = 0; r < 8; ++r) {
            float vv = acc3[r];
            vv = 1.f / (1.f + __expf(-vv));
            out[(size_t)(gbase + r + 8*hlf) * 3 + m] = vv;
        }
    }
    if (m == 0) {                // opticity column held by lanes {0, 16}
        #pragma unroll
        for (int r = 0; r < 8; ++r) {
            float vv = accw[r];
            vv = 1.f / (1.f + __expf(-vv));
            out[(size_t)NPTS * 3 + gbase + r + 8*hlf] = vv;
        }
    }
}

extern "C" void kernel_launch(void* const* d_in, const int* in_sizes, int n_in,
                              void* d_out, int out_size, void* d_ws, size_t ws_size,
                              hipStream_t stream) {
    (void)in_sizes; (void)n_in; (void)out_size; (void)ws_size;
    const float* x     = (const float*)d_in[0];
    const float* d     = (const float*)d_in[1];
    const float* table = (const float*)d_in[2];
    const float* Wv0   = (const float*)d_in[3];
    const float* Wv1   = (const float*)d_in[4];
    const float* Wv2   = (const float*)d_in[5];
    const float* Ws0   = (const float*)d_in[6];
    const float* Ws1   = (const float*)d_in[7];
    const int*   deg   = (const int*)d_in[8];
    half_t* wsh = (half_t*)d_ws;
    float*  out = (float*)d_out;

    auto pack = [&](const float* W, half_t* o, int Kr, int Nr, int KT, int NT, int hklim) {
        int total = KT * NT * 512;
        ngp_pack_w<<<(total + 255) / 256, 256, 0, stream>>>(W, o, Kr, Nr, KT, NT, hklim);
    };
    pack(Wv0, wsh + OFF_V0, 512, 64, 16, 4, 512); // all feat rows carry h -> scale 1/256
    pack(Wv1, wsh + OFF_V1,  64, 64,  2, 4, 0);
    pack(Wv2, wsh + OFF_V2,  64,  3,  2, 1, 0);
    pack(Ws0, wsh + OFF_S0,  40, 64,  2, 4, 32);  // rows 0..31 are h -> scale 1/256
    pack(Ws1, wsh + OFF_S1,  64,  1,  2, 1, 0);

    ResCfg rc;
    const double scale = std::exp(std::log(4096.0 / 16.0) / 15.0);
    for (int l = 0; l < 16; ++l)
        rc.r[l] = (int)std::ceil(16.0 * std::pow(scale, (double)l));

    ngp_fused_kernel<<<NPTS / 128, 256, 0, stream>>>(x, d, table, deg, wsh, out, rc);
}
